// CrossModalAttention_747324309856
// MI455X (gfx1250) — compile-verified, matrix operand order checked by reading, not run on hardware
//
#include <hip/hip_runtime.h>
#include <hip/hip_bf16.h>

typedef __bf16 bf16_t;
typedef __attribute__((ext_vector_type(16))) __bf16 v16bf;
typedef __attribute__((ext_vector_type(8)))  float  v8f;
typedef __attribute__((ext_vector_type(4)))  unsigned int u32x4;
typedef __attribute__((ext_vector_type(8)))  int          i32x8;
typedef __attribute__((ext_vector_type(4)))  int          i32x4;

#define BM 128
#define BN 128
#define BK 32
#define LDS_STRIDE 48   // bf16 elems per row: 32 data + 16 pad -> 96B rows (16B aligned)

enum { EPI_PROJ = 0, EPI_PROJ_T = 1, EPI_SCORES = 2, EPI_OUT = 3 };

union FragBF {
  v16bf v;
  uint4 u[2];
};

union Pack4 {
  bf16_t h[4];
  uint2  u;
};

union Pack8 {
  bf16_t h[8];
  uint4  u;
};

union Bits16 {
  bf16_t h;
  unsigned short s;
};

__device__ __forceinline__ v8f wmma_bf16(const v16bf a, const v16bf b, const v8f c) {
  return __builtin_amdgcn_wmma_f32_16x16x32_bf16(false, a, false, b, (short)0, c, false, false);
}

// ---------------- Tensor Data Mover: 128x32 bf16 tile, global -> padded LDS ----------------
// D# group0: count=1 | lds_addr | 57b global_addr | type=2
// D# group1: data_size=2B, pad_enable, pad_interval=3 (16 DWORDs = one 64B row),
//            pad_amount=7 (8 DWORDs = 32B)  => LDS row stride 96B == LDS_STRIDE*2
__device__ __forceinline__ void tdm_load_tile_bf16(unsigned ldsAddr,
                                                   const unsigned short* gsrc,
                                                   int ld, int rows) {
  const unsigned long long ga = (unsigned long long)(uintptr_t)gsrc;
  u32x4 g0 = {0u, 0u, 0u, 0u};
  g0[0] = 1u;                                                  // count=1 (valid), user mode
  g0[1] = ldsAddr;                                             // LDS byte address
  g0[2] = (unsigned)ga;                                        // global_addr[31:0]
  g0[3] = (unsigned)((ga >> 32) & 0x01ffffffull) | (2u << 30); // addr[56:32] | type=2
  i32x8 g1 = {0, 0, 0, 0, 0, 0, 0, 0};
  g1[0] = (1 << 16) | (1 << 20) | (3 << 22) | (7 << 25);       // data_size/pad cfg
  g1[1] = (int)((unsigned)ld << 16);                           // tensor_dim0[15:0]
  g1[2] = (int)(((unsigned)rows << 16) | ((unsigned)ld >> 16));// dim1[15:0] | dim0[31:16]
  g1[3] = (int)((32u << 16) | ((unsigned)rows >> 16));         // tile_dim0=32 | dim1 hi
  g1[4] = 128;                                                 // tile_dim1 = 128 rows
  g1[5] = ld;                                                  // tensor_dim0_stride lo32
  const i32x4 gz4 = {0, 0, 0, 0};
  const i32x8 gz8 = {0, 0, 0, 0, 0, 0, 0, 0};
  // 6-arg toolchain variant: (g0, g1, g2, g3, extra, cpol)
  __builtin_amdgcn_tensor_load_to_lds(g0, g1, gz4, gz4, gz8, 0);
}

// ---------------- VGPR path: 128x32 f32 tile -> convert bf16 -> padded LDS ----------------
__device__ __forceinline__ void load_tile_f32(const float* __restrict__ src, int ld,
                                              int rowBase, int kBase,
                                              unsigned short* __restrict__ sm, int tid) {
#pragma unroll
  for (int i = 0; i < 4; ++i) {
    int linear = tid + i * 256;          // 0..1023
    int r  = linear >> 3;
    int c4 = linear & 7;
    const float4 f = *(const float4*)(src + (size_t)(rowBase + r) * ld + kBase + c4 * 4);
    Pack4 p;
    p.h[0] = (bf16_t)f.x; p.h[1] = (bf16_t)f.y;
    p.h[2] = (bf16_t)f.z; p.h[3] = (bf16_t)f.w;
    *(uint2*)(&sm[r * LDS_STRIDE + c4 * 4]) = p.u;
  }
}

// Per-lane A/B fragment per CDNA5 16-bit 16x32 layout.
__device__ __forceinline__ FragBF load_frag(const unsigned short* __restrict__ sm,
                                            int rowBase, int lane) {
  FragBF f;
  int row  = rowBase + (lane & 15);
  int koff = (lane & 16) ? 8 : 0;
  const unsigned short* p = sm + row * LDS_STRIDE + koff;
  f.u[0] = *(const uint4*)(p);
  f.u[1] = *(const uint4*)(p + 16);
  return f;
}

// C = A(MxK) * B(NxK)^T, K contiguous in both. 128x128x32 tiles, 8 waves (4Mx2N),
// each wave computes 32x64 via 2x4 WMMA accumulators.
template <bool F32SRC, int EPI>
__global__ __launch_bounds__(256)
void gemm_bf16_wmma(const void* __restrict__ Av, const void* __restrict__ Bv,
                    const float* __restrict__ bias,
                    void* __restrict__ Cv, const float* __restrict__ Xres,
                    long aBatch, long bBatch, long cBatch,
                    int M, int N, int K, int Sy, int Dd) {
  __shared__ __align__(16) unsigned short smA[2][BM * LDS_STRIDE];
  __shared__ __align__(16) unsigned short smB[2][BN * LDS_STRIDE];

  const int tid   = threadIdx.x;
  const int lane  = tid & 31;
  const int w     = tid >> 5;
  const int waveM = (w & 3) * 32;
  const int waveN = (w >> 2) * 64;
  const int bm    = blockIdx.x * BM;
  const int bn    = blockIdx.y * BN;
  const int z     = blockIdx.z;

  const v8f zero8 = {0.f, 0.f, 0.f, 0.f, 0.f, 0.f, 0.f, 0.f};
  v8f acc[2][4];
#pragma unroll
  for (int i = 0; i < 2; ++i)
#pragma unroll
    for (int j = 0; j < 4; ++j) acc[i][j] = zero8;

  const int kIters = K / BK;

  if constexpr (F32SRC) {
    // ---- VGPR convert path (projections), single LDS buffer ----
    const float* Af = (const float*)Av + (size_t)z * aBatch;
    const float* Bf = (const float*)Bv + (size_t)z * bBatch;
#pragma unroll 1
    for (int kt = 0; kt < kIters; ++kt) {
      const int k0 = kt * BK;
      __syncthreads();
      load_tile_f32(Af, K, bm, k0, smA[0], tid);
      load_tile_f32(Bf, K, bn, k0, smB[0], tid);
      if (kt + 1 < kIters && tid < BM) {
        __builtin_prefetch((const char*)(Af + (size_t)(bm + tid) * K + k0 + BK), 0, 0);
        __builtin_prefetch((const char*)(Bf + (size_t)(bn + tid) * K + k0 + BK), 0, 0);
      }
      __syncthreads();
      FragBF a0 = load_frag(smA[0], waveM, lane);
      FragBF a1 = load_frag(smA[0], waveM + 16, lane);
      FragBF bfr[4];
#pragma unroll
      for (int ns = 0; ns < 4; ++ns) bfr[ns] = load_frag(smB[0], waveN + ns * 16, lane);
#pragma unroll
      for (int ns = 0; ns < 4; ++ns) {
        acc[0][ns] = wmma_bf16(a0.v, bfr[ns].v, acc[0][ns]);
        acc[1][ns] = wmma_bf16(a1.v, bfr[ns].v, acc[1][ns]);
      }
    }
  } else {
    // ---- TDM async path (scores / PV), double buffered ----
    const unsigned short* Ah = (const unsigned short*)Av + (size_t)z * aBatch;
    const unsigned short* Bh = (const unsigned short*)Bv + (size_t)z * bBatch;
    const unsigned ldsA0 = (unsigned)(uintptr_t)(void*)&smA[0][0];
    const unsigned ldsA1 = (unsigned)(uintptr_t)(void*)&smA[1][0];
    const unsigned ldsB0 = (unsigned)(uintptr_t)(void*)&smB[0][0];
    const unsigned ldsB1 = (unsigned)(uintptr_t)(void*)&smB[1][0];

    if (w == 0) {  // prologue: DMA tile 0 into buffer 0
      tdm_load_tile_bf16(ldsA0, Ah + (size_t)bm * K, K, M);
      tdm_load_tile_bf16(ldsB0, Bh + (size_t)bn * K, K, N);
    }
#pragma unroll 1
    for (int kt = 0; kt < kIters; ++kt) {
      const int cur = kt & 1;
      __syncthreads();                       // prev compute done -> other buffer reusable
      if (w == 0) {
        __builtin_amdgcn_s_wait_tensorcnt(0);  // tile kt resident in LDS
        if (kt + 1 < kIters) {                 // overlap: DMA tile kt+1 during compute kt
          const int k1 = (kt + 1) * BK;
          tdm_load_tile_bf16(cur ? ldsA0 : ldsA1, Ah + (size_t)bm * K + k1, K, M);
          tdm_load_tile_bf16(cur ? ldsB0 : ldsB1, Bh + (size_t)bn * K + k1, K, N);
        }
      }
      __syncthreads();                       // publish tile kt to all waves

      FragBF a0 = load_frag(smA[cur], waveM, lane);
      FragBF a1 = load_frag(smA[cur], waveM + 16, lane);
      FragBF bfr[4];
#pragma unroll
      for (int ns = 0; ns < 4; ++ns) bfr[ns] = load_frag(smB[cur], waveN + ns * 16, lane);
#pragma unroll
      for (int ns = 0; ns < 4; ++ns) {
        acc[0][ns] = wmma_bf16(a0.v, bfr[ns].v, acc[0][ns]);
        acc[1][ns] = wmma_bf16(a1.v, bfr[ns].v, acc[1][ns]);
      }
    }
  }

  // Epilogue. C fragment layout: lane<16 -> M=i, lane>=16 -> M=8+i; N = lane&15.
  const int n0    = lane & 15;
  const int mhalf = (lane >> 4) * 8;
#pragma unroll
  for (int ms = 0; ms < 2; ++ms) {
#pragma unroll
    for (int ns = 0; ns < 4; ++ns) {
      const int gn = bn + waveN + ns * 16 + n0;
      float bval = 0.f;
      if constexpr (EPI == EPI_PROJ || EPI == EPI_PROJ_T) bval = bias[gn];

      if constexpr (EPI == EPI_PROJ_T) {
        // Transposed store: the 8 fragment rows are contiguous along sy -> one b128 store.
        const int gmBase = bm + waveM + ms * 16 + mhalf;
        const int b  = gmBase / Sy;
        const int sy = gmBase - b * Sy;
        Pack8 pk;
#pragma unroll
        for (int i = 0; i < 8; ++i) pk.h[i] = (bf16_t)(acc[ms][ns][i] + bval);
        *(uint4*)&((unsigned short*)Cv)[((size_t)b * Dd + gn) * Sy + sy] = pk.u;
      } else {
#pragma unroll
        for (int i = 0; i < 8; ++i) {
          const int gm = bm + waveM + ms * 16 + mhalf + i;
          float val = acc[ms][ns][i] + bval;
          if constexpr (EPI == EPI_PROJ) {
            Bits16 c; c.h = (bf16_t)val;
            ((unsigned short*)Cv)[(size_t)gm * N + gn] = c.s;
          } else if constexpr (EPI == EPI_SCORES) {
            ((float*)Cv)[(size_t)z * cBatch + (size_t)gm * N + gn] = val;
          } else { // EPI_OUT: residual add, f32 out
            const size_t idx = (size_t)z * cBatch + (size_t)gm * N + gn;
            ((float*)Cv)[idx] = val + Xres[idx];
          }
        }
      }
    }
  }
}

// Row softmax with mask; one block per (b, sx) row of Sy=2048, 8 elems/thread.
__global__ __launch_bounds__(256)
void softmax_mask(const float* __restrict__ scores, const int* __restrict__ mask,
                  unsigned short* __restrict__ P, int Sy) {
  const size_t row = blockIdx.x;
  const float* s = scores + row * (size_t)Sy;
  const int*   m = mask   + row * (size_t)Sy;
  unsigned short* p = P + row * (size_t)Sy;
  const int tid  = threadIdx.x;
  const int lane = tid & 31;
  const int w    = tid >> 5;

  float v[8];
  float mx = -3.4e38f;
#pragma unroll
  for (int i = 0; i < 8; ++i) {
    const int j = tid + i * 256;
    float x = s[j];
    x = (m[j] == 0) ? -1e9f : x;
    v[i] = x;
    mx = fmaxf(mx, x);
  }
#pragma unroll
  for (int o = 16; o > 0; o >>= 1) mx = fmaxf(mx, __shfl_xor(mx, o, 32));

  __shared__ float redMax[8];
  __shared__ float redSum[8];
  if (lane == 0) redMax[w] = mx;
  __syncthreads();
  float bm = redMax[0];
#pragma unroll
  for (int i = 1; i < 8; ++i) bm = fmaxf(bm, redMax[i]);

  float sum = 0.f;
#pragma unroll
  for (int i = 0; i < 8; ++i) {
    v[i] = __expf(v[i] - bm);
    sum += v[i];
  }
#pragma unroll
  for (int o = 16; o > 0; o >>= 1) sum += __shfl_xor(sum, o, 32);
  if (lane == 0) redSum[w] = sum;
  __syncthreads();
  float tot = 0.f;
#pragma unroll
  for (int i = 0; i < 8; ++i) tot += redSum[i];
  const float inv = 1.0f / tot;

#pragma unroll
  for (int i = 0; i < 8; ++i) {
    const int j = tid + i * 256;
    Bits16 c; c.h = (bf16_t)(v[i] * inv);
    p[j] = c.s;
  }
}

extern "C" void kernel_launch(void* const* d_in, const int* in_sizes, int n_in,
                              void* d_out, int out_size, void* d_ws, size_t ws_size,
                              hipStream_t stream) {
  const float* x   = (const float*)d_in[0];
  const float* y   = (const float*)d_in[1];
  const int*   msk = (const int*)d_in[2];
  const float* Wq  = (const float*)d_in[3];
  const float* bq  = (const float*)d_in[4];
  const float* Wk  = (const float*)d_in[5];
  const float* bk  = (const float*)d_in[6];
  const float* Wv  = (const float*)d_in[7];
  const float* bv  = (const float*)d_in[8];
  float* out = (float*)d_out;

  const int Bn = 4, S = 2048, D = 1024;
  const size_t MB = 1024ull * 1024ull;

  char* ws = (char*)d_ws;
  unsigned short* q  = (unsigned short*)(ws);              // 16 MB bf16 [B*S, D]
  unsigned short* kk = (unsigned short*)(ws + 16 * MB);    // 16 MB bf16 [B*S, D]
  unsigned short* vT = (unsigned short*)(ws + 32 * MB);    // 16 MB bf16 [B, D, S]
  float*          sc = (float*)(ws + 48 * MB);             // 64 MB f32  [B, S, S]
  unsigned short* P  = (unsigned short*)(ws + 112 * MB);   // 32 MB bf16 [B, S, S]

  dim3 blk(256);

  // Projections: M = B*S = 8192, N = K = D = 1024
  gemm_bf16_wmma<true, EPI_PROJ><<<dim3(64, 8, 1), blk, 0, stream>>>(
      x, Wq, bq, q, nullptr, 0, 0, 0, Bn * S, D, D, S, D);
  gemm_bf16_wmma<true, EPI_PROJ><<<dim3(64, 8, 1), blk, 0, stream>>>(
      y, Wk, bk, kk, nullptr, 0, 0, 0, Bn * S, D, D, S, D);
  gemm_bf16_wmma<true, EPI_PROJ_T><<<dim3(64, 8, 1), blk, 0, stream>>>(
      y, Wv, bv, vT, nullptr, 0, 0, 0, Bn * S, D, D, S, D);

  // Scores: per batch q[b] (SxD) * k[b]^T -> f32 [S,S]  (TDM-staged tiles)
  gemm_bf16_wmma<false, EPI_SCORES><<<dim3(16, 16, 4), blk, 0, stream>>>(
      q, kk, nullptr, sc, nullptr,
      (long)S * D, (long)S * D, (long)S * S, S, S, D, S, D);

  // Masked softmax rows -> P bf16
  softmax_mask<<<dim3(Bn * S), blk, 0, stream>>>(sc, msk, P, S);

  // Out: per batch P[b] (SxS) * vT[b]^T (DxS) -> f32 [S,D], + x residual (TDM-staged)
  gemm_bf16_wmma<false, EPI_OUT><<<dim3(16, 8, 4), blk, 0, stream>>>(
      P, vT, nullptr, out, x,
      (long)S * S, (long)D * S, (long)S * D, S, D, S, S, D);
}
